// LiftSplatShoot_mask_mix_3375844294853
// MI455X (gfx1250) — compile-verified, matrix operand order checked by reading, not run on hardware
//
#include <hip/hip_runtime.h>
#include <hip/hip_bf16.h>
#include <math.h>

typedef __attribute__((ext_vector_type(16))) _Float16 v16h;
typedef __attribute__((ext_vector_type(8)))  float    v8f;

#define BB   4
#define NN   6
#define CCH  64
#define DD   41
#define FH   16
#define FW   44
#define FHW  (FH*FW)            // 704
#define NPIX (BB*NN*FH*FW)      // 16896
#define NPTS (BB*NN*DD*FH*FW)   // 692736
#define NX0  200
#define NX1  200
#define NVOXB (NX0*NX1)         // 40000 per (b,c) plane
#define OUTN (BB*CCH*NX0*NX1)   // 10,240,000 floats
#define DPAD 48

// workspace layout (byte offsets)
#define WS_MATS_OFF  0                                   // 24 cams * 24 floats
#define WS_DEPTH_OFF 4096                                // NPIX * DPAD floats
#define WS_VOX_OFF   (4096 + (size_t)NPIX*DPAD*4)        // NPTS int32

__device__ inline void inv3x3(const float m[9], float inv[9]) {
    float a00=m[0],a01=m[1],a02=m[2],a10=m[3],a11=m[4],a12=m[5],a20=m[6],a21=m[7],a22=m[8];
    float det = a00*(a11*a22-a12*a21) - a01*(a10*a22-a12*a20) + a02*(a10*a21-a11*a20);
    float r = 1.0f/det;
    inv[0]=(a11*a22-a12*a21)*r; inv[1]=(a02*a21-a01*a22)*r; inv[2]=(a01*a12-a02*a11)*r;
    inv[3]=(a12*a20-a10*a22)*r; inv[4]=(a00*a22-a02*a20)*r; inv[5]=(a02*a10-a00*a12)*r;
    inv[6]=(a10*a21-a11*a20)*r; inv[7]=(a01*a20-a00*a21)*r; inv[8]=(a00*a11-a01*a10)*r;
}

// one thread per camera: invPost[9], combine[9], trans[3], post_trans[3]
__global__ void lss_prep_mats(const float* __restrict__ rots,
                              const float* __restrict__ trans,
                              const float* __restrict__ intrins,
                              const float* __restrict__ post_rots,
                              const float* __restrict__ post_trans,
                              float* __restrict__ mats) {
    int t = threadIdx.x;
    if (t >= BB*NN) return;
    float pr[9], ipr[9], K[9], iK[9], R[9], Cmb[9];
    #pragma unroll
    for (int i = 0; i < 9; ++i) { pr[i]=post_rots[t*9+i]; K[i]=intrins[t*9+i]; R[i]=rots[t*9+i]; }
    inv3x3(pr, ipr);
    inv3x3(K, iK);
    #pragma unroll
    for (int i = 0; i < 3; ++i)
        #pragma unroll
        for (int j = 0; j < 3; ++j)
            Cmb[i*3+j] = R[i*3+0]*iK[0*3+j] + R[i*3+1]*iK[1*3+j] + R[i*3+2]*iK[2*3+j];
    float* o = mats + t*24;
    #pragma unroll
    for (int i = 0; i < 9; ++i) { o[i]=ipr[i]; o[9+i]=Cmb[i]; }
    #pragma unroll
    for (int i = 0; i < 3; ++i) { o[18+i]=trans[t*3+i]; o[21+i]=post_trans[t*3+i]; }
}

__global__ void lss_zero_out(float* __restrict__ out) {
    int i = blockIdx.x * blockDim.x + threadIdx.x;
    if (i < OUTN) out[i] = 0.0f;
}

// softmax over D per pixel; write [pix][DPAD] padded with zeros
__global__ void lss_softmax_depth(const float* __restrict__ feat_depth,
                                  float* __restrict__ depth_ws) {
    int p = blockIdx.x * blockDim.x + threadIdx.x;
    if (p >= NPIX) return;
    int b   = p / (NN*FHW);
    int rem = p - b*(NN*FHW);
    int n   = rem / FHW;
    int pix = rem - n*FHW;
    const float* src = feat_depth + ((size_t)(b*NN+n)*DD)*FHW + pix;
    float mx = -INFINITY;
    for (int d = 0; d < DD; ++d) { float v = src[d*FHW]; mx = v > mx ? v : mx; }
    float s = 0.0f;
    float e[DD];
    for (int d = 0; d < DD; ++d) { e[d] = expf(src[d*FHW] - mx); s += e[d]; }
    float rs = 1.0f / s;
    float* dst = depth_ws + (size_t)p*DPAD;
    for (int d = 0; d < DD; ++d) dst[d] = e[d]*rs;
    for (int d = DD; d < DPAD; ++d) dst[d] = 0.0f;
}

// per frustum point: voxel base index (b*C*NX0*NX1-folded) or -1 if invalid
__global__ void lss_geom_vox(const float* __restrict__ mats,
                             int* __restrict__ vox_ws) {
    int i = blockIdx.x * blockDim.x + threadIdx.x;
    if (i >= NPTS) return;
    int b    = i / (NN*DD*FHW);
    int rem  = i - b*(NN*DD*FHW);
    int n    = rem / (DD*FHW);
    int rem2 = rem - n*(DD*FHW);
    int d    = rem2 / FHW;
    int pix  = rem2 - d*FHW;
    int h = pix / FW, w = pix - h*FW;
    const float* m = mats + (b*NN + n)*24;

    // frustum pixel (u,v,depth): linspace(0,703,44), linspace(0,255,16), 4+d
    float u   = (float)w * (703.0f/43.0f);
    float v   = (float)h * (255.0f/15.0f);
    float dep = 4.0f + (float)d;
    // undo post-augmentation
    float px = u   - m[21];
    float py = v   - m[22];
    float pz = dep - m[23];
    float q0 = m[0]*px + m[1]*py + m[2]*pz;
    float q1 = m[3]*px + m[4]*py + m[5]*pz;
    float q2 = m[6]*px + m[7]*py + m[8]*pz;
    // un-project (u*d, v*d, d), then combine + trans
    float r0 = q0*q2, r1 = q1*q2, r2 = q2;
    float gx = m[ 9]*r0 + m[10]*r1 + m[11]*r2 + m[18];
    float gy = m[12]*r0 + m[13]*r1 + m[14]*r2 + m[19];
    float gz = m[15]*r0 + m[16]*r1 + m[17]*r2 + m[20];
    // (geom - (bx - dx/2)) / dx, truncation toward zero like .astype(int32)
    int xi = (int)((gx + 50.0f) / 0.5f);
    int yi = (int)((gy + 50.0f) / 0.5f);
    int zi = (int)((gz + 10.0f) / 20.0f);
    bool valid = (xi >= 0) & (xi < NX0) & (yi >= 0) & (yi < NX1) & (zi == 0);
    vox_ws[i] = valid ? (b*(CCH*NVOXB) + xi*NX1 + yi) : -1;
}

// one wave per pixel: 3x4 WMMA tiles of the rank-1 lift, scatter via f32 atomics
__global__ __launch_bounds__(256) void lss_splat_wmma(const float* __restrict__ feat_pv,
                                                      const float* __restrict__ depth_ws,
                                                      const int*   __restrict__ vox_ws,
                                                      float* __restrict__ out) {
    const int wave = (int)(threadIdx.x >> 5);
    const int lane = (int)(threadIdx.x & 31);
    const int p = blockIdx.x * 8 + wave;
    if (p >= NPIX) return;

    int b   = p / (NN*FHW);
    int rem = p - b*(NN*FHW);
    int n   = rem / FHW;
    int pix = rem - n*FHW;
    const int cam = b*NN + n;

    const float* fbase = feat_pv  + (size_t)cam*CCH*FHW + pix;
    const float* dbase = depth_ws + (size_t)p*DPAD;
    const int*   vbase = vox_ws   + (size_t)cam*DD*FHW + pix;
    __builtin_prefetch(fbase, 0, 1);

    const int  l15 = lane & 15;
    const bool lo  = lane < 16;

    // per-channel-chunk feature halves (B matrix, K=0 row)
    _Float16 bh[4];
    #pragma unroll
    for (int cc = 0; cc < 4; ++cc) {
        float fv = fbase[(size_t)(cc*16 + l15) * FHW];
        bh[cc] = lo ? (_Float16)fv : (_Float16)0.0f;
    }

    #pragma unroll
    for (int dc = 0; dc < 3; ++dc) {
        float dv = dbase[dc*16 + l15];            // padded: rows 41..47 are zero
        _Float16 ah = lo ? (_Float16)dv : (_Float16)0.0f;

        // voxel base for each of this lane's 8 accumulator rows (M = dc*16 + r [+8 for hi half])
        int vb[8];
        #pragma unroll
        for (int r = 0; r < 8; ++r) {
            int mrow = dc*16 + r + (lo ? 0 : 8);
            vb[r] = (mrow < DD) ? vbase[(size_t)mrow * FHW] : -1;
        }

        #pragma unroll
        for (int cc = 0; cc < 4; ++cc) {
            v16h A, Bm;
            #pragma unroll
            for (int i = 0; i < 16; ++i) { A[i] = (_Float16)0.0f; Bm[i] = (_Float16)0.0f; }
            A[0]  = ah;         // A[m][k=0] = depth chunk, lanes 0-15
            Bm[0] = bh[cc];     // B[k=0][n] = feature chunk, lanes 0-15
            v8f acc;
            #pragma unroll
            for (int i = 0; i < 8; ++i) acc[i] = 0.0f;
            acc = __builtin_amdgcn_wmma_f32_16x16x32_f16(
                      false, A, false, Bm, (short)0, acc, false, false);

            const int cidx = cc*16 + l15;
            float* ocol = out + (size_t)cidx * NVOXB;   // out[((b*C + c)*200+x)*200+y]
            #pragma unroll
            for (int r = 0; r < 8; ++r) {
                if (vb[r] >= 0) atomicAdd(&ocol[vb[r]], acc[r]);
            }
        }
    }
}

extern "C" void kernel_launch(void* const* d_in, const int* in_sizes, int n_in,
                              void* d_out, int out_size, void* d_ws, size_t ws_size,
                              hipStream_t stream) {
    const float* feat_depth = (const float*)d_in[0];
    const float* feat_pv    = (const float*)d_in[1];
    const float* rots       = (const float*)d_in[2];
    const float* trans      = (const float*)d_in[3];
    const float* intrins    = (const float*)d_in[4];
    const float* post_rots  = (const float*)d_in[5];
    const float* post_trans = (const float*)d_in[6];
    float* out = (float*)d_out;

    char* ws = (char*)d_ws;
    float* mats     = (float*)(ws + WS_MATS_OFF);
    float* depth_ws = (float*)(ws + WS_DEPTH_OFF);
    int*   vox_ws   = (int*)  (ws + WS_VOX_OFF);

    lss_prep_mats<<<1, 32, 0, stream>>>(rots, trans, intrins, post_rots, post_trans, mats);
    lss_zero_out<<<(OUTN + 255)/256, 256, 0, stream>>>(out);
    lss_softmax_depth<<<(NPIX + 255)/256, 256, 0, stream>>>(feat_depth, depth_ws);
    lss_geom_vox<<<(NPTS + 255)/256, 256, 0, stream>>>(mats, vox_ws);
    lss_splat_wmma<<<NPIX/8, 256, 0, stream>>>(feat_pv, depth_ws, vox_ws, out);
}
